// EncdecMultiheadAttn_1786706395345
// MI455X (gfx1250) — matrix operand order repaired and checked by reading, not compiled, not run on hardware
//
#include <hip/hip_runtime.h>
#include <hip/hip_bf16.h>

// Problem constants (fixed by the reference harness)
#define LQ 1024
#define LK 1024
#define BB 8
#define DD 1024
#define HH 16
#define DKH 64   // head dim

typedef unsigned short u16;
typedef __attribute__((ext_vector_type(16))) __bf16 v16bf;
typedef __attribute__((ext_vector_type(8)))  float  v8f;

union FragB {
    v16bf v;
    uint4 u[2];
    u16   h[16];
};

__device__ __forceinline__ u16 f2bf(float f) {
    unsigned u = __float_as_uint(f);
    unsigned r = 0x7FFFu + ((u >> 16) & 1u);   // round-to-nearest-even
    return (u16)((u + r) >> 16);
}

// ---------------------------------------------------------------------------
// LayerNorm over last dim (D=1024), one block per row, bf16 output
// ---------------------------------------------------------------------------
__global__ void __launch_bounds__(256)
ln_bf16_kernel(const float* __restrict__ q, const float* __restrict__ g,
               const float* __restrict__ beta, u16* __restrict__ qn) {
    const int row = blockIdx.x;
    const int tid = threadIdx.x;
    const float4 x = ((const float4*)(q + (long)row * DD))[tid];
    float s  = x.x + x.y + x.z + x.w;
    float ss = x.x * x.x + x.y * x.y + x.z * x.z + x.w * x.w;
    __shared__ float sh0[256], sh1[256];
    sh0[tid] = s; sh1[tid] = ss;
    __syncthreads();
    for (int o = 128; o > 0; o >>= 1) {
        if (tid < o) { sh0[tid] += sh0[tid + o]; sh1[tid] += sh1[tid + o]; }
        __syncthreads();
    }
    const float mu  = sh0[0] * (1.0f / DD);
    const float var = sh1[0] * (1.0f / DD) - mu * mu;
    const float rs  = rsqrtf(var + 1e-5f);
    const float4 gg = ((const float4*)g)[tid];
    const float4 bb = ((const float4*)beta)[tid];
    ushort4 o4;
    o4.x = f2bf((x.x - mu) * rs * gg.x + bb.x);
    o4.y = f2bf((x.y - mu) * rs * gg.y + bb.y);
    o4.z = f2bf((x.z - mu) * rs * gg.z + bb.z);
    o4.w = f2bf((x.w - mu) * rs * gg.w + bb.w);
    ((ushort4*)(qn + (long)row * DD))[tid] = o4;
}

// ---------------------------------------------------------------------------
// fp32 -> bf16 elementwise (n4 = count/4)
// ---------------------------------------------------------------------------
__global__ void __launch_bounds__(256)
cvt_bf16_kernel(const float* __restrict__ in, u16* __restrict__ out, long n4) {
    const long i = (long)blockIdx.x * 256 + threadIdx.x;
    if (i >= n4) return;
    const float4 v = ((const float4*)in)[i];
    ushort4 o4;
    o4.x = f2bf(v.x); o4.y = f2bf(v.y); o4.z = f2bf(v.z); o4.w = f2bf(v.w);
    ((ushort4*)out)[i] = o4;
}

// ---------------------------------------------------------------------------
// LDS-staged batched WMMA GEMM:  C[z] = alpha * A[z] * W[z]^T (+ Res)
//   A: [M,K] row-major; bf16, or fp32 converted to bf16 during LDS staging.
//   W: [N,K] row-major bf16.   C: fp32 (optional), Cb: bf16 (optional).
//   batch z: z0 = z / zdiv, z1 = z % zdiv; offsets/strides in elements.
// Block = 8 waves arranged WMW x WNW; each wave owns a 32x32 tile (2x2 WMMA).
// bf16-A path: double-buffered GLOBAL_LOAD_ASYNC_TO_LDS_B128 pipeline
// (ASYNCcnt-tracked DMA, overlapped with WMMA; one barrier per K-step).
// fp32-A path: synchronous staged convert (used for attn @ V).
// ---------------------------------------------------------------------------
template <int WMW, int WNW, bool AF32>
__global__ void __launch_bounds__(256)
gemm_bf16_wmma(const void* __restrict__ Av, long sA, long bA0, long bA1,
               const u16* __restrict__ W, long sW, long bW0, long bW1,
               float* __restrict__ C, long sC, long bC0, long bC1,
               u16* __restrict__ Cb, long sCb, long bCb0, long bCb1,
               const float* __restrict__ Res,
               int K, int zdiv, float alpha) {
    constexpr int TM = WMW * 32;      // block tile rows (A)
    constexpr int TN = WNW * 32;      // block tile cols (W rows)
    constexpr int PITCH = 40;         // LDS row pitch in bf16 elems (32 + 8 pad)
    constexpr int NB = AF32 ? 1 : 2;  // double-buffer only the async path
    static_assert((TM * 4) % 256 == 0, "A tile chunks must divide 256 threads");
    __shared__ u16 lsA[NB][TM * PITCH];
    __shared__ u16 lsW[NB][TN * PITCH];

    const int z  = blockIdx.z;
    const int z0 = z / zdiv, z1 = z % zdiv;
    const long coff  = (long)z0 * bC0  + (long)z1 * bC1;
    const long cboff = (long)z0 * bCb0 + (long)z1 * bCb1;

    const int tid  = threadIdx.x;
    const int wave = tid >> 5;
    const int lane = tid & 31;
    const int wm = wave / WNW;
    const int wn = wave % WNW;
    const int ml = lane & 15;         // M row (A frag) / N col (B frag)
    const int kh = lane >> 4;         // K-half selector

    const int m0 = blockIdx.y * TM;
    const int n0 = blockIdx.x * TN;

    const float* Af = nullptr;
    const u16*   Ah = nullptr;
    if constexpr (AF32) Af = (const float*)Av + (long)z0 * bA0 + (long)z1 * bA1 + (long)m0 * sA;
    else                Ah = (const u16*)Av   + (long)z0 * bA0 + (long)z1 * bA1 + (long)m0 * sA;
    const u16* Wb = W + (long)z0 * bW0 + (long)z1 * bW1 + (long)n0 * sW;

    v8f acc00 = {}, acc01 = {}, acc10 = {}, acc11 = {};

    // ---- fragment reads from LDS (conflict-free, 16B each) + 4 WMMAs ----
    auto compute = [&](const u16* bufA, const u16* bufW) {
        const u16* la0 = &bufA[(wm * 32 + ml)      * PITCH];
        const u16* la1 = &bufA[(wm * 32 + 16 + ml) * PITCH];
        const u16* lb0 = &bufW[(wn * 32 + ml)      * PITCH];
        const u16* lb1 = &bufW[(wn * 32 + 16 + ml) * PITCH];
        FragB a0, a1, b0, b1;
        a0.u[0] = *(const uint4*)(la0 + kh * 8);       // A: K = kh*8..+7
        a0.u[1] = *(const uint4*)(la0 + 16 + kh * 8);  //    K = 16+kh*8..+7
        a1.u[0] = *(const uint4*)(la1 + kh * 8);
        a1.u[1] = *(const uint4*)(la1 + 16 + kh * 8);
        b0.u[0] = *(const uint4*)(lb0 + kh * 8);       // B: K = kh*8..+7, 16+kh*8..+7
        b0.u[1] = *(const uint4*)(lb0 + 16 + kh * 8);
        b1.u[0] = *(const uint4*)(lb1 + kh * 8);
        b1.u[1] = *(const uint4*)(lb1 + 16 + kh * 8);
        acc00 = __builtin_amdgcn_wmma_f32_16x16x32_bf16(false, a0.v, false, b0.v,
                                                        (short)0, acc00, false, false);
        acc01 = __builtin_amdgcn_wmma_f32_16x16x32_bf16(false, a0.v, false, b1.v,
                                                        (short)0, acc01, false, false);
        acc10 = __builtin_amdgcn_wmma_f32_16x16x32_bf16(false, a1.v, false, b0.v,
                                                        (short)0, acc10, false, false);
        acc11 = __builtin_amdgcn_wmma_f32_16x16x32_bf16(false, a1.v, false, b1.v,
                                                        (short)0, acc11, false, false);
    };

    if constexpr (!AF32) {
        static_assert((TN * 4) % 256 == 0, "W tile chunks must divide 256 threads");
        // Async memory->LDS staging of one K-tile (16B chunk per thread slot).
        auto stage = [&](int k0, u16* bA, u16* bW) {
#pragma unroll
            for (int i = 0; i < (TM * 4) / 256; ++i) {
                const int idx = tid + i * 256;
                const int r = idx >> 2, c = (idx & 3) * 8;
                const unsigned dst = (unsigned)(size_t)(&bA[r * PITCH + c]);
                const u16* src = Ah + (long)r * sA + k0 + c;
                asm volatile("global_load_async_to_lds_b128 %0, %1, off"
                             :: "v"(dst), "v"(src) : "memory");
            }
#pragma unroll
            for (int i = 0; i < (TN * 4) / 256; ++i) {
                const int idx = tid + i * 256;
                const int r = idx >> 2, c = (idx & 3) * 8;
                const unsigned dst = (unsigned)(size_t)(&bW[r * PITCH + c]);
                const u16* src = Wb + (long)r * sW + k0 + c;
                asm volatile("global_load_async_to_lds_b128 %0, %1, off"
                             :: "v"(dst), "v"(src) : "memory");
            }
        };

        stage(0, lsA[0], lsW[0]);
        asm volatile("s_wait_asynccnt 0x0" ::: "memory");
        __syncthreads();
        const int T = K >> 5;
        for (int i = 0; i < T; ++i) {
            if (i + 1 < T) stage((i + 1) << 5, lsA[(i + 1) & 1], lsW[(i + 1) & 1]);
            compute(lsA[i & 1], lsW[i & 1]);
            // wait own async writes (tile i+1) landed, then join all waves:
            // barrier => everyone done reading buf[i&1] AND tile i+1 visible.
            asm volatile("s_wait_asynccnt 0x0" ::: "memory");
            __syncthreads();
        }
    } else {
        // Synchronous staging with fp32 -> bf16 conversion.
        for (int k0 = 0; k0 < K; k0 += 32) {
#pragma unroll
            for (int i = 0; i < (TM * 4) / 256; ++i) {
                const int idx = tid + i * 256;
                const int r = idx >> 2, c = (idx & 3) * 8;
                const float* p = Af + (long)r * sA + k0 + c;
                ushort4 lo; lo.x = f2bf(p[0]); lo.y = f2bf(p[1]); lo.z = f2bf(p[2]); lo.w = f2bf(p[3]);
                ushort4 hi; hi.x = f2bf(p[4]); hi.y = f2bf(p[5]); hi.z = f2bf(p[6]); hi.w = f2bf(p[7]);
                *(ushort4*)&lsA[0][r * PITCH + c]     = lo;
                *(ushort4*)&lsA[0][r * PITCH + c + 4] = hi;
                if (k0 + 32 < K) __builtin_prefetch(p + 32, 0, 1);
            }
            {
                constexpr int WTOT = TN * 4;
                constexpr int WIT  = (WTOT + 255) / 256;
#pragma unroll
                for (int i = 0; i < WIT; ++i) {
                    const int idx = tid + i * 256;
                    if ((WTOT % 256) == 0 || idx < WTOT) {
                        const int r = idx >> 2, c = (idx & 3) * 8;
                        *(uint4*)&lsW[0][r * PITCH + c] =
                            *(const uint4*)(Wb + (long)r * sW + k0 + c);
                    }
                }
            }
            __syncthreads();
            compute(lsA[0], lsW[0]);
            __syncthreads();
        }
    }

    auto store_tile = [&](const v8f& acc, int mt, int nt) {
        const int nc = nt + ml;
#pragma unroll
        for (int r = 0; r < 8; ++r) {
            const int m = mt + r + kh * 8;   // C layout: VGPR r -> M=r (+8 hi lanes)
            float v = acc[r] * alpha;
            if (Res) v += Res[(long)m * sC + nc];
            if (C)  C [coff  + (long)m * sC  + nc] = v;
            if (Cb) Cb[cboff + (long)m * sCb + nc] = f2bf(v);
        }
    };
    store_tile(acc00, m0 + wm * 32,      n0 + wn * 32);
    store_tile(acc01, m0 + wm * 32,      n0 + wn * 32 + 16);
    store_tile(acc10, m0 + wm * 32 + 16, n0 + wn * 32);
    store_tile(acc11, m0 + wm * 32 + 16, n0 + wn * 32 + 16);
}

// ---------------------------------------------------------------------------
// Masked softmax over rows of attn (in place, fp32). One block per row.
// row = (b*H + h)*LQ + q ; columns = LK
// ---------------------------------------------------------------------------
__global__ void __launch_bounds__(256)
softmax_kernel(float* __restrict__ attn, const unsigned char* __restrict__ mask) {
    const long row = blockIdx.x;
    const int  b   = (int)(row / ((long)HH * LQ));
    const int  tid = threadIdx.x;
    float* p = attn + row * LK;
    float4 v = ((float4*)p)[tid];
    const uchar4 mk = ((const uchar4*)(mask + (long)b * LK))[tid];
    if (mk.x) v.x = -1e9f;
    if (mk.y) v.y = -1e9f;
    if (mk.z) v.z = -1e9f;
    if (mk.w) v.w = -1e9f;

    __shared__ float sh[256];
    float mx = fmaxf(fmaxf(v.x, v.y), fmaxf(v.z, v.w));
    sh[tid] = mx; __syncthreads();
    for (int o = 128; o > 0; o >>= 1) {
        if (tid < o) sh[tid] = fmaxf(sh[tid], sh[tid + o]);
        __syncthreads();
    }
    mx = sh[0];
    __syncthreads();
    v.x = __expf(v.x - mx); v.y = __expf(v.y - mx);
    v.z = __expf(v.z - mx); v.w = __expf(v.w - mx);
    sh[tid] = v.x + v.y + v.z + v.w; __syncthreads();
    for (int o = 128; o > 0; o >>= 1) {
        if (tid < o) sh[tid] += sh[tid + o];
        __syncthreads();
    }
    const float inv = 1.0f / sh[0];
    v.x *= inv; v.y *= inv; v.z *= inv; v.w *= inv;
    ((float4*)p)[tid] = v;
}

// ---------------------------------------------------------------------------
// Transpose V (from KV bf16, [kk][b][2D] layout, V at +D) into Vt[b][h][n][kk]
// ---------------------------------------------------------------------------
__global__ void __launch_bounds__(256)
transpose_v_kernel(const u16* __restrict__ KV, u16* __restrict__ Vt) {
    const long i = (long)blockIdx.x * 256 + threadIdx.x;   // B*H*DKH*LK total
    const int kk = (int)(i & (LK - 1));
    long t = i >> 10;
    const int n = (int)(t & (DKH - 1)); t >>= 6;
    const int h = (int)(t & (HH - 1));
    const int b = (int)(t >> 4);
    Vt[i] = KV[((long)kk * BB + b) * (2 * DD) + DD + h * DKH + n];
}

// ---------------------------------------------------------------------------
extern "C" void kernel_launch(void* const* d_in, const int* in_sizes, int n_in,
                              void* d_out, int out_size, void* d_ws, size_t ws_size,
                              hipStream_t stream) {
    (void)in_sizes; (void)n_in; (void)out_size; (void)ws_size;
    const float* q    = (const float*)d_in[0];
    const float* k    = (const float*)d_in[1];
    const unsigned char* mask = (const unsigned char*)d_in[2];
    const float* Wq   = (const float*)d_in[3];
    const float* Wkv  = (const float*)d_in[4];
    const float* Wo   = (const float*)d_in[5];
    const float* ln_g = (const float*)d_in[6];
    const float* ln_b = (const float*)d_in[7];

    float* out  = (float*)d_out;                       // [LQ*B*D]
    float* attn = out + (size_t)LQ * BB * DD;          // [B*H*LQ*LK]

    char* ws = (char*)d_ws;
    size_t off = 0;
    auto wsalloc = [&](size_t bytes) -> void* {
        void* p = ws + off; off += (bytes + 255) & ~(size_t)255; return p;
    };
    u16* qn_bf  = (u16*)wsalloc((size_t)LQ * BB * DD * 2);       // 16 MB
    u16* k_bf   = (u16*)wsalloc((size_t)LK * BB * DD * 2);       // 16 MB
    u16* Wq_bf  = (u16*)wsalloc((size_t)DD * DD * 2);            //  2 MB
    u16* Wkv_bf = (u16*)wsalloc((size_t)2 * DD * DD * 2);        //  4 MB
    u16* Wo_bf  = (u16*)wsalloc((size_t)DD * DD * 2);            //  2 MB
    u16* Q_bf   = (u16*)wsalloc((size_t)LQ * BB * DD * 2);       // 16 MB
    u16* KV_bf  = (u16*)wsalloc((size_t)LK * BB * 2 * DD * 2);   // 32 MB
    u16* Vt_bf  = (u16*)wsalloc((size_t)BB * HH * DKH * LK * 2); // 16 MB
    u16* ctx_bf = (u16*)wsalloc((size_t)LQ * BB * DD * 2);       // 16 MB

    const dim3 blk(256);

    // 1) LayerNorm(q) -> qn (bf16)
    ln_bf16_kernel<<<dim3(LQ * BB), blk, 0, stream>>>(q, ln_g, ln_b, qn_bf);

    // 2) fp32 -> bf16 conversions
    cvt_bf16_kernel<<<dim3((LK * BB * DD / 4) / 256), blk, 0, stream>>>(k, k_bf, LK * BB * DD / 4);
    cvt_bf16_kernel<<<dim3((DD * DD / 4) / 256), blk, 0, stream>>>(Wq, Wq_bf, DD * DD / 4);
    cvt_bf16_kernel<<<dim3((2 * DD * DD / 4) / 256), blk, 0, stream>>>(Wkv, Wkv_bf, 2 * DD * DD / 4);
    cvt_bf16_kernel<<<dim3((DD * DD / 4) / 256), blk, 0, stream>>>(Wo, Wo_bf, DD * DD / 4);

    // 3) Q = qn @ Wq^T * dk^-0.5   (M=8192, N=1024, K=1024) -> bf16
    gemm_bf16_wmma<2, 4, false><<<dim3(DD / 128, LQ * BB / 64, 1), blk, 0, stream>>>(
        qn_bf, DD, 0, 0,
        Wq_bf, DD, 0, 0,
        nullptr, 0, 0, 0,
        Q_bf, DD, 0, 0,
        nullptr, DD, 1, 0.125f);

    // 4) KV = k @ Wkv^T            (M=8192, N=2048, K=1024) -> bf16
    gemm_bf16_wmma<2, 4, false><<<dim3(2 * DD / 128, LK * BB / 64, 1), blk, 0, stream>>>(
        k_bf, DD, 0, 0,
        Wkv_bf, DD, 0, 0,
        nullptr, 0, 0, 0,
        KV_bf, 2 * DD, 0, 0,
        nullptr, DD, 1, 1.0f);

    // 5) scores[b,h] = Q_h K_h^T  (z = b*H + h; M=LQ, N=LK, K=dk) -> attn (fp32)
    gemm_bf16_wmma<2, 4, false><<<dim3(LK / 128, LQ / 64, BB * HH), blk, 0, stream>>>(
        Q_bf, (long)BB * DD, DD, DKH,
        KV_bf, (long)BB * 2 * DD, 2 * DD, DKH,             // K part of KV
        attn, LK, (long)HH * LQ * LK, (long)LQ * LK,
        nullptr, 0, 0, 0,
        nullptr, DKH, HH, 1.0f);

    // 6) masked softmax over attn rows (in place)
    softmax_kernel<<<dim3(BB * HH * LQ), blk, 0, stream>>>(attn, mask);

    // 7) V transpose -> Vt[b][h][n][kk]
    transpose_v_kernel<<<dim3(BB * HH * DKH * LK / 256), blk, 0, stream>>>(KV_bf, Vt_bf);

    // 8) ctx[b,h] = attn @ V_h  (A fp32 staged->bf16; 8x1 waves, N=dk) -> ctx bf16
    gemm_bf16_wmma<8, 1, true><<<dim3(DKH / 32, LQ / 256, BB * HH), blk, 0, stream>>>(
        attn, LK, (long)HH * LQ * LK, (long)LQ * LK,
        Vt_bf, LK, (long)HH * DKH * LK, (long)DKH * LK,
        nullptr, 0, 0, 0,
        ctx_bf, (long)BB * DD, DD, DKH,
        nullptr, LK, HH, 1.0f);

    // 9) out = ctx @ Wo^T + residual(q)   (M=8192, N=1024, K=1024) -> fp32
    gemm_bf16_wmma<2, 4, false><<<dim3(DD / 128, LQ * BB / 64, 1), blk, 0, stream>>>(
        ctx_bf, DD, 0, 0,
        Wo_bf, DD, 0, 0,
        out, DD, 0, 0,
        nullptr, 0, 0, 0,
        q, DD, 1, 1.0f);
}
